// MultiHeadProbSparseAttention_64707977281897
// MI455X (gfx1250) — matrix operand order repaired and checked by reading, not compiled, not run on hardware
//
#include <hip/hip_runtime.h>
#include <hip/hip_bf16.h>
#include <math.h>

// ---------------------------------------------------------------------------
// Problem constants (from reference): B=4, T=S=2048, D=512, H=8, DK=DV=64
// sample_k = u = int(4*ln(2048)) = 30
// ---------------------------------------------------------------------------
#define Bn   4
#define Tn   2048
#define Sn   2048
#define Dn   512
#define Hn   8
#define DKn  64
#define DVn  64
#define HDn  512     // H*DK == H*DV
#define SKn  30      // sample_k
#define Un   30      // top-u

typedef __attribute__((ext_vector_type(16))) __bf16 v16bf;
typedef __attribute__((ext_vector_type(8)))  __bf16 v8bf;
typedef __attribute__((ext_vector_type(8)))  float  v8f;
typedef __attribute__((ext_vector_type(4)))  float  v4f;
typedef __attribute__((ext_vector_type(4)))  int    v4i;

#if defined(__HIP_DEVICE_COMPILE__) && __has_builtin(__builtin_amdgcn_global_load_async_to_lds_b128)
#define USE_ASYNC_LDS 1
#else
#define USE_ASYNC_LDS 0
#endif

#if USE_ASYNC_LDS
__device__ __forceinline__ void async_copy16(const __bf16* g, __bf16* l) {
    // GLOBAL_LOAD_ASYNC_TO_LDS_B128: (global <4 x i32>* src, lds <4 x i32>* dst,
    //                                 imm offset, cpol)
    __builtin_amdgcn_global_load_async_to_lds_b128(
        (__attribute__((address_space(1))) v4i*)g,
        (__attribute__((address_space(3))) v4i*)l, 0, 0);
}
__device__ __forceinline__ void wait_async0() {
#if __has_builtin(__builtin_amdgcn_s_wait_asynccnt)
    __builtin_amdgcn_s_wait_asynccnt(0);
#else
    asm volatile("s_wait_asynccnt 0x0" ::: "memory");
#endif
}
#endif

// ---------------------------------------------------------------------------
// Wave32 reductions
// ---------------------------------------------------------------------------
__device__ __forceinline__ float warpReduceSum(float v) {
#pragma unroll
    for (int o = 16; o > 0; o >>= 1) v += __shfl_xor(v, o, 32);
    return v;
}
__device__ __forceinline__ float warpReduceMax(float v) {
#pragma unroll
    for (int o = 16; o > 0; o >>= 1) v = fmaxf(v, __shfl_xor(v, o, 32));
    return v;
}
__device__ __forceinline__ float blockReduceSum(float v, float* sh) {
    int lane = threadIdx.x & 31, w = threadIdx.x >> 5;
    v = warpReduceSum(v);
    if (lane == 0) sh[w] = v;
    __syncthreads();
    float r = (threadIdx.x < 8) ? sh[threadIdx.x] : 0.0f;
    if (w == 0) { r = warpReduceSum(r); if (lane == 0) sh[0] = r; }
    __syncthreads();
    float out = sh[0];
    __syncthreads();
    return out;
}
__device__ __forceinline__ float blockReduceMax(float v, float* sh) {
    int lane = threadIdx.x & 31, w = threadIdx.x >> 5;
    v = warpReduceMax(v);
    if (lane == 0) sh[w] = v;
    __syncthreads();
    float r = (threadIdx.x < 8) ? sh[threadIdx.x] : -1e30f;
    if (w == 0) { r = warpReduceMax(r); if (lane == 0) sh[0] = r; }
    __syncthreads();
    float out = sh[0];
    __syncthreads();
    return out;
}

// ---------------------------------------------------------------------------
// fp32 -> bf16 elementwise convert
// ---------------------------------------------------------------------------
__global__ void f32_to_bf16_kernel(const float* __restrict__ in,
                                   __bf16* __restrict__ out, size_t n) {
    size_t i = (size_t)blockIdx.x * blockDim.x + threadIdx.x;
    size_t stride = (size_t)gridDim.x * blockDim.x;
    for (; i < n; i += stride) out[i] = (__bf16)in[i];
}

// W (KxN fp32, row-major) -> Wt (NxK bf16, row-major == column-major of W)
__global__ void transpose_w_bf16_kernel(const float* __restrict__ W,
                                        __bf16* __restrict__ Wt, int K, int N) {
    int idx = blockIdx.x * blockDim.x + threadIdx.x;
    if (idx >= K * N) return;
    int nn = idx / K, kk = idx % K;
    Wt[(size_t)nn * K + kk] = (__bf16)W[(size_t)kk * N + nn];
}

// ---------------------------------------------------------------------------
// bf16 WMMA GEMM: C[M,N] (fp32) = A[M,K] (bf16 row-major) * Bt[N,K]^T + bias
// Block = 256 threads (8 waves stacked along M), each wave one 16x16 tile via
// v_wmma_f32_16x16x32_bf16. The 16-column B tile is shared by all 8 waves:
// staged once into LDS with GLOBAL_LOAD_ASYNC_TO_LDS_B128 (ASYNCcnt), then
// fragments come from conflict-free ds_load_b128 (+8 bf16 column pad).
// Grid = (N/16, M/128). Requires Kr % 32 == 0, Kr <= 512.
// ---------------------------------------------------------------------------
__global__ __launch_bounds__(256)
void gemm_bf16_wmma_kernel(const __bf16* __restrict__ A,
                           const __bf16* __restrict__ Bt,
                           const float*  __restrict__ bias,
                           float* __restrict__ C,
                           int Mr, int Nr, int Kr) {
    __shared__ alignas(16) __bf16 ldsB[16 * (512 + 8)];

    const int tid   = threadIdx.x;
    const int lane  = tid & 31;
    const int wave  = tid >> 5;
    const int tileM = blockIdx.y * 128 + wave * 16;
    const int tileN = blockIdx.x * 16;
    const int hl    = lane >> 4;      // lane half (0/1)
    const int l16   = lane & 15;
    const int ldsStride = Kr + 8;     // bf16 elements per staged column (padded)

    // ---- stage B tile: 16 cols x Kr bf16, 16B chunks, async DMA to LDS ----
    {
        const int col = tid & 15;                     // 16 threads per column
        const int cpc = Kr >> 3;                      // 16B chunks per column
        const __bf16* gcol = Bt + (size_t)(tileN + col) * Kr;
        __bf16* lcol = &ldsB[col * ldsStride];
        for (int j = (tid >> 4); j < cpc; j += 16) {
#if USE_ASYNC_LDS
            async_copy16(gcol + j * 8, lcol + j * 8);
#else
            *(v8bf*)(lcol + j * 8) = *(const v8bf*)(gcol + j * 8);
#endif
        }
#if USE_ASYNC_LDS
        wait_async0();
#endif
        __syncthreads();
    }

    // A fragment: row = tileM+l16; K elems {hl*8+0..7} and {16+hl*8+0..7}
    const __bf16* ap = A + (size_t)(tileM + l16) * Kr + hl * 8;
    // B fragment from LDS: col = l16; K elems {hl*8+0..7} and {16+hl*8+0..7}
    const __bf16* bp = &ldsB[l16 * ldsStride + hl * 8];

    v8f acc = {};
    for (int k0 = 0; k0 < Kr; k0 += 32) {
        __builtin_prefetch(ap + 128, 0, 1);   // global_prefetch_b8
        v8bf alo = *(const v8bf*)(ap);
        v8bf ahi = *(const v8bf*)(ap + 16);
        v8bf blo = *(const v8bf*)(bp + k0);       // ds_load_b128
        v8bf bhi = *(const v8bf*)(bp + k0 + 16);  // ds_load_b128
        v16bf a, b;
#pragma unroll
        for (int i = 0; i < 8; ++i) {
            a[i] = alo[i]; a[i + 8] = ahi[i];
            b[i] = blo[i]; b[i + 8] = bhi[i];
        }
        acc = __builtin_amdgcn_wmma_f32_16x16x32_bf16(
            /*neg_a=*/false, a, /*neg_b=*/false, b,
            /*c_mod=*/(short)0, acc, /*reuse_a=*/false, /*reuse_b=*/false);
        ap += 32;
    }

    const int col  = tileN + l16;
    const float bv = bias ? bias[col] : 0.0f;
    // D layout: VGPR j -> row tileM + j + hl*8, col = tileN + l16
    float* cp = C + (size_t)(tileM + hl * 8) * Nr + col;
#pragma unroll
    for (int j = 0; j < 8; ++j) cp[(size_t)j * Nr] = acc[j] + bv;
}

// ---------------------------------------------------------------------------
// Fill with constant: final_scores = 1/S. Pure streaming -> nontemporal
// stores (TH=NT) so the 537MB fill doesn't rinse L2 (k/v reuse lives there).
// ---------------------------------------------------------------------------
__global__ void fill_f32v4_kernel(v4f* __restrict__ p, float val, size_t n4) {
    size_t i = (size_t)blockIdx.x * blockDim.x + threadIdx.x;
    size_t stride = (size_t)gridDim.x * blockDim.x;
    v4f v = {val, val, val, val};
    for (; i < n4; i += stride) __builtin_nontemporal_store(v, p + i);
}

// ---------------------------------------------------------------------------
// Sampled QK -> M = max_j - mean_j over 30 sampled keys. One wave per head.
// grid = B*T blocks of 256 (8 waves == 8 heads). M layout: (B*H, T).
// ---------------------------------------------------------------------------
__global__ __launch_bounds__(256)
void qk_sample_kernel(const float* __restrict__ q, const float* __restrict__ k,
                      const int* __restrict__ index_sample,
                      float* __restrict__ Mbuf) {
    int b = blockIdx.x / Tn;
    int t = blockIdx.x % Tn;
    int h = threadIdx.x >> 5;
    int lane = threadIdx.x & 31;

    const float* qp = q + ((size_t)(b * Tn + t)) * HDn + h * DKn + lane * 2;
    float q0 = qp[0], q1 = qp[1];

    float mx = -1e30f, sm = 0.0f;
#pragma unroll 1
    for (int j = 0; j < SKn; ++j) {
        int sidx = index_sample[t * SKn + j];
        const float* kp = k + ((size_t)(b * Sn + sidx)) * HDn + h * DKn + lane * 2;
        float p = q0 * kp[0] + q1 * kp[1];
        p = warpReduceSum(p);
        mx = fmaxf(mx, p);
        sm += p;
    }
    if (lane == 0)
        Mbuf[((size_t)(b * Hn + h)) * Tn + t] = mx - sm * (1.0f / SKn);
}

// ---------------------------------------------------------------------------
// Iterative top-U argmax per (b,h) row of M. Deterministic tie-break: lower t.
// grid = B*H blocks of 256. Destroys Mbuf (rebuilt each launch).
// ---------------------------------------------------------------------------
__global__ __launch_bounds__(256)
void topk_kernel(float* __restrict__ Mbuf, int* __restrict__ top) {
    __shared__ float sv[256];
    __shared__ int   si[256];
    float* Mv = Mbuf + (size_t)blockIdx.x * Tn;
    int*   tp = top + blockIdx.x * Un;
    int tid = threadIdx.x;

    for (int it = 0; it < Un; ++it) {
        float best = -1e30f;
        int   bi   = Tn;
        for (int t = tid; t < Tn; t += 256) {
            float vv = Mv[t];
            if (vv > best || (vv == best && t < bi)) { best = vv; bi = t; }
        }
        sv[tid] = best; si[tid] = bi;
        __syncthreads();
        for (int s2 = 128; s2 > 0; s2 >>= 1) {
            if (tid < s2) {
                if (sv[tid + s2] > sv[tid] ||
                    (sv[tid + s2] == sv[tid] && si[tid + s2] < si[tid])) {
                    sv[tid] = sv[tid + s2]; si[tid] = si[tid + s2];
                }
            }
            __syncthreads();
        }
        if (tid == 0) { tp[it] = si[0]; Mv[si[0]] = -1e30f; }
        __syncthreads();
    }
}

// ---------------------------------------------------------------------------
// v mean over S per (b, channel). grid*block = B*HDn threads.
// ---------------------------------------------------------------------------
__global__ void v_mean_kernel(const float* __restrict__ v, float* __restrict__ vm) {
    int idx = blockIdx.x * blockDim.x + threadIdx.x;
    if (idx >= Bn * HDn) return;
    int b = idx / HDn, c = idx % HDn;
    const float* p = v + (size_t)b * Sn * HDn + c;
    float s = 0.0f;
    for (int j = 0; j < Sn; ++j) s += p[(size_t)j * HDn];
    vm[idx] = s * (1.0f / Sn);
}

// ctx[b,t,c] = vmean[b,c]
__global__ void ctx_fill_kernel(float* __restrict__ ctx,
                                const float* __restrict__ vm, size_t n) {
    size_t i = (size_t)blockIdx.x * blockDim.x + threadIdx.x;
    size_t stride = (size_t)gridDim.x * blockDim.x;
    for (; i < n; i += stride) {
        int c = (int)(i % HDn);
        int b = (int)(i / ((size_t)Tn * HDn));
        ctx[i] = vm[b * HDn + c];
    }
}

// ---------------------------------------------------------------------------
// Full attention for one (b,h,u): scores over S, softmax, scatter scores row
// into final_scores, attn = scores @ v scattered into ctx.
// grid = B*H*U blocks of 256.
// ---------------------------------------------------------------------------
__global__ __launch_bounds__(256)
void attn_kernel(const float* __restrict__ q, const float* __restrict__ k,
                 const float* __restrict__ v, const int* __restrict__ top,
                 float* __restrict__ fscores, float* __restrict__ ctx) {
    __shared__ float qsh[DKn];
    __shared__ float red[8];
    __shared__ float accsh[DVn];

    int task = blockIdx.x;
    int u = task % Un;
    int h = (task / Un) % Hn;
    int b = task / (Un * Hn);
    int tid  = threadIdx.x;
    int lane = tid & 31;
    int trow = top[(b * Hn + h) * Un + u];

    if (tid < DKn) {
        qsh[tid]   = q[((size_t)(b * Tn + trow)) * HDn + h * DKn + tid];
        accsh[tid] = 0.0f;
    }
    __syncthreads();

    float loc[8];
    float lmax = -1e30f;
#pragma unroll 1
    for (int i = 0; i < 8; ++i) {
        int s = i * 256 + tid;
        const float* kp = k + ((size_t)(b * Sn + s)) * HDn + h * DKn;
        float dot = 0.0f;
#pragma unroll
        for (int d = 0; d < DKn; ++d) dot += qsh[d] * kp[d];
        loc[i] = dot * 0.125f;   // 1/sqrt(64)
        lmax = fmaxf(lmax, loc[i]);
    }
    float bmax = blockReduceMax(lmax, red);

    float lsum = 0.0f;
#pragma unroll
    for (int i = 0; i < 8; ++i) { loc[i] = __expf(loc[i] - bmax); lsum += loc[i]; }
    float bsum = blockReduceSum(lsum, red);
    float inv = 1.0f / bsum;

    float* srow = fscores + (((size_t)(b * Hn + h)) * Tn + trow) * (size_t)Sn;
#pragma unroll
    for (int i = 0; i < 8; ++i) {
        int s = i * 256 + tid;
        loc[i] *= inv;
        srow[s] = loc[i];
    }

    // attn[d] = sum_s loc[s] * v[b,s,h,d]
#pragma unroll 1
    for (int d = 0; d < DVn; ++d) {
        float p = 0.0f;
#pragma unroll
        for (int i = 0; i < 8; ++i) {
            int s = i * 256 + tid;
            p += loc[i] * v[((size_t)(b * Sn + s)) * HDn + h * DVn + d];
        }
        p = warpReduceSum(p);
        if (lane == 0) atomicAdd(&accsh[d], p);
    }
    __syncthreads();
    if (tid < DVn)
        ctx[((size_t)(b * Tn + trow)) * HDn + h * DVn + tid] = accsh[tid];
}

// ---------------------------------------------------------------------------
// Launch
// ---------------------------------------------------------------------------
extern "C" void kernel_launch(void* const* d_in, const int* in_sizes, int n_in,
                              void* d_out, int out_size, void* d_ws, size_t ws_size,
                              hipStream_t stream) {
    const float* x_q = (const float*)d_in[0];
    const float* x_k = (const float*)d_in[1];
    const float* x_v = (const float*)d_in[2];
    const float* Wq  = (const float*)d_in[3];
    const float* bq  = (const float*)d_in[4];
    const float* Wk  = (const float*)d_in[5];
    const float* bk  = (const float*)d_in[6];
    const float* Wv  = (const float*)d_in[7];
    const float* bv  = (const float*)d_in[8];
    const float* Wo  = (const float*)d_in[9];
    const float* bo  = (const float*)d_in[10];
    const int*   idx = (const int*)d_in[11];

    const size_t BT = (size_t)Bn * Tn;                 // 8192 rows
    const size_t nAct = BT * Dn;                       // 4,194,304

    // workspace carve-out (256B aligned)
    size_t off = 0;
    auto alloc = [&](size_t bytes) -> void* {
        void* p = (char*)d_ws + off;
        off = (off + bytes + 255) & ~(size_t)255;
        return p;
    };
    __bf16* xq_bf  = (__bf16*)alloc(nAct * 2);
    __bf16* xk_bf  = (__bf16*)alloc(nAct * 2);
    __bf16* xv_bf  = (__bf16*)alloc(nAct * 2);
    __bf16* ctx_bf = (__bf16*)alloc(nAct * 2);
    __bf16* Wqt = (__bf16*)alloc((size_t)Dn * HDn * 2);
    __bf16* Wkt = (__bf16*)alloc((size_t)Dn * HDn * 2);
    __bf16* Wvt = (__bf16*)alloc((size_t)Dn * HDn * 2);
    __bf16* Wot = (__bf16*)alloc((size_t)HDn * HDn * 2);
    float* qf   = (float*)alloc(nAct * 4);
    float* kf   = (float*)alloc(nAct * 4);
    float* vf   = (float*)alloc(nAct * 4);
    float* ctxf = (float*)alloc(nAct * 4);
    float* Mbuf = (float*)alloc((size_t)Bn * Hn * Tn * 4);
    int*   topb = (int*)alloc((size_t)Bn * Hn * Un * 4);
    float* vme  = (float*)alloc((size_t)Bn * HDn * 4);

    float* out     = (float*)d_out;                       // (B,T,H*DV)
    float* fscores = out + BT * HDn;                      // (B,H,T,S)

    // 1. activations fp32 -> bf16
    f32_to_bf16_kernel<<<4096, 256, 0, stream>>>(x_q, xq_bf, nAct);
    f32_to_bf16_kernel<<<4096, 256, 0, stream>>>(x_k, xk_bf, nAct);
    f32_to_bf16_kernel<<<4096, 256, 0, stream>>>(x_v, xv_bf, nAct);

    // 2. weights: transpose + convert
    int wElems = Dn * HDn;
    transpose_w_bf16_kernel<<<(wElems + 255) / 256, 256, 0, stream>>>(Wq, Wqt, Dn, HDn);
    transpose_w_bf16_kernel<<<(wElems + 255) / 256, 256, 0, stream>>>(Wk, Wkt, Dn, HDn);
    transpose_w_bf16_kernel<<<(wElems + 255) / 256, 256, 0, stream>>>(Wv, Wvt, Dn, HDn);
    transpose_w_bf16_kernel<<<(wElems + 255) / 256, 256, 0, stream>>>(Wo, Wot, HDn, HDn);

    // 3. projections via WMMA GEMM
    dim3 gg(HDn / 16, (int)(BT / 128));   // (32, 64)
    gemm_bf16_wmma_kernel<<<gg, 256, 0, stream>>>(xq_bf, Wqt, bq, qf, (int)BT, HDn, Dn);
    gemm_bf16_wmma_kernel<<<gg, 256, 0, stream>>>(xk_bf, Wkt, bk, kf, (int)BT, HDn, Dn);
    gemm_bf16_wmma_kernel<<<gg, 256, 0, stream>>>(xv_bf, Wvt, bv, vf, (int)BT, HDn, Dn);

    // 4. final_scores = 1/S (nontemporal streaming fill; attn overwrites rows)
    size_t n4 = ((size_t)Bn * Hn * Tn * Sn) / 4;
    fill_f32v4_kernel<<<8192, 256, 0, stream>>>((v4f*)fscores, 1.0f / Sn, n4);

    // 5. sampled QK -> M
    qk_sample_kernel<<<Bn * Tn, 256, 0, stream>>>(qf, kf, idx, Mbuf);

    // 6. top-u per (b,h)
    topk_kernel<<<Bn * Hn, 256, 0, stream>>>(Mbuf, topb);

    // 7. v mean + ctx broadcast fill
    v_mean_kernel<<<(Bn * HDn + 255) / 256, 256, 0, stream>>>(vf, vme);
    ctx_fill_kernel<<<4096, 256, 0, stream>>>(ctxf, vme, nAct);

    // 8. dense attention on selected queries, scatter into fscores + ctx
    attn_kernel<<<Bn * Hn * Un, 256, 0, stream>>>(qf, kf, vf, topb, fscores, ctxf);

    // 9. output projection: ctx -> bf16 -> WMMA GEMM (+bo) -> out
    f32_to_bf16_kernel<<<4096, 256, 0, stream>>>(ctxf, ctx_bf, nAct);
    gemm_bf16_wmma_kernel<<<gg, 256, 0, stream>>>(ctx_bf, Wot, bo, out, (int)BT, HDn, HDn);
}
